// GATLayer_15710990369229
// MI455X (gfx1250) — matrix operand order repaired and checked by reading, not compiled, not run on hardware
//
#include <hip/hip_runtime.h>
#include <hip/hip_bf16.h>
#include <math.h>

typedef __attribute__((ext_vector_type(16))) _Float16 v16h;
typedef __attribute__((ext_vector_type(8)))  float    v8f;

static constexpr int Bc = 2, Sc = 512, Dc = 768, Hc = 12, HDc = 64, Ec = 50, DEc = 50, Fc = 128;

// ---- workspace layout (float offsets) ----
static constexpr long FEAT_OFF  = 0;                                  // [B*S, D]
static constexpr long ATTN_OFF  = FEAT_OFF  + (long)Bc * Sc * Dc;     // [B,H,S,S]
static constexpr long GCNA_OFF  = ATTN_OFF  + (long)Bc * Hc * Sc * Sc;// [B*S, D]
static constexpr long GCNP_OFF  = GCNA_OFF  + (long)Bc * Sc * Dc;     // [B*S, D]
static constexpr long SRC_OFF   = GCNP_OFF  + (long)Bc * Sc * Dc;     // [B,H,S]
static constexpr long DST_OFF   = SRC_OFF   + (long)Bc * Hc * Sc;     // [B,H,S]
static constexpr long EDIAG_OFF = DST_OFF   + (long)Bc * Hc * Sc;     // [B*S, E]
static constexpr long N1_OFF    = EDIAG_OFF + (long)Bc * Sc * Ec;     // [B*S, DE]
static constexpr long N2_OFF    = N1_OFF    + (long)Bc * Sc * DEc;
static constexpr long EI_OFF    = N2_OFF    + (long)Bc * Sc * DEc;
static constexpr long EJ_OFF    = EI_OFF    + (long)Bc * Sc * DEc;
static constexpr long P_OFF     = EJ_OFF    + (long)Bc * Sc * DEc;
static constexpr long Q_OFF     = P_OFF     + (long)Bc * Sc * DEc;

// ---------------------------------------------------------------------------
// WMMA fragment loaders from LDS, in this code's contiguous k order (A and B match).
// A (16x32 f16): lane m=l&15, hi=l>>4; half i holds K = i + 16*hi
// B (32x16 f16): lane n=l&15, hi=l>>4; half i holds K = i + 16*hi
// C/D (16x16 f32): vgpr r holds M = r + 8*hi, N = l&15
// ---------------------------------------------------------------------------
__device__ __forceinline__ v16h load_a_frag(const _Float16* Am, int ld) {
  const int lane = threadIdx.x & 31;
  const int m = lane & 15, hi = (lane >> 4) & 1;
  const _Float16* base = Am + m * ld;
  union { v16h h; unsigned u[8]; } r;
#pragma unroll
  for (int p = 0; p < 8; ++p) {
    const int k = 2 * p + (hi << 4);
    r.u[p] = *(const unsigned*)(base + k);
  }
  return r.h;
}

__device__ __forceinline__ v16h load_b_frag(const _Float16* Bm, int ld) {
  const int lane = threadIdx.x & 31;
  const int n = lane & 15, hi = (lane >> 4) & 1;
  const unsigned* p = (const unsigned*)(Bm + n * ld + (hi << 4));
  union { v16h h; unsigned u[8]; } r;
#pragma unroll
  for (int i = 0; i < 8; ++i) r.u[i] = p[i];
  return r.h;
}

#define WMMA_F32_F16(a, b, c) \
  __builtin_amdgcn_wmma_f32_16x16x32_f16(false, (a), false, (b), (short)0, (c), false, false)

// convert float4 -> 4 x f16, one ds_store_b64
__device__ __forceinline__ void cvt_store4(_Float16* dst, float4 v) {
  union { _Float16 h[4]; unsigned long long u; } t;
  t.h[0] = (_Float16)v.x; t.h[1] = (_Float16)v.y;
  t.h[2] = (_Float16)v.z; t.h[3] = (_Float16)v.w;
  *(unsigned long long*)dst = t.u;
}

// ---------------------------------------------------------------------------
// Fully in-bounds GEMM: C[.,64x64 tile] = A[M,K] @ W[N,K]^T + bias.
// Requires M%64==0, N%64==0, K%32==0, lda/ldw%4==0. Software-pipelined:
// regs hold chunk k+32 while WMMA consumes LDS chunk k.
// ---------------------------------------------------------------------------
__global__ __launch_bounds__(256) void gemm_full_kernel(
    const float* __restrict__ A, int lda,
    const float* __restrict__ W, int ldw,
    const float* __restrict__ bias,
    float* __restrict__ C, int ldc, int K) {
  constexpr int KP = 40;
  __attribute__((aligned(16))) __shared__ _Float16 sA[64 * KP];
  __attribute__((aligned(16))) __shared__ _Float16 sB[64 * KP];
  const int tid = threadIdx.x, lane = tid & 31, w = tid >> 5;
  const int rowBlk = w & 3, colHalf = w >> 2;
  const int m0 = blockIdx.y * 64, n0 = blockIdx.x * 64;
  const int hi = lane >> 4, nl = lane & 15;

  // staging: 64 rows x 32 k = 512 float4; 2 per thread per tensor
  const int r0 = tid >> 3, kq = (tid & 7) * 4;  // rows 0..31 ; +32 for t=1
  const float* pA0 = A + (long)(m0 + r0) * lda + kq;
  const float* pA1 = A + (long)(m0 + r0 + 32) * lda + kq;
  const float* pB0 = W + (long)(n0 + r0) * ldw + kq;
  const float* pB1 = W + (long)(n0 + r0 + 32) * ldw + kq;

  float4 a0 = *(const float4*)pA0;
  float4 a1 = *(const float4*)pA1;
  float4 b0 = *(const float4*)pB0;
  float4 b1 = *(const float4*)pB1;

  v8f acc0 = {}, acc1 = {};
  for (int k0 = 0; k0 < K; k0 += 32) {
    __syncthreads();
    cvt_store4(&sA[r0 * KP + kq], a0);
    cvt_store4(&sA[(r0 + 32) * KP + kq], a1);
    cvt_store4(&sB[r0 * KP + kq], b0);
    cvt_store4(&sB[(r0 + 32) * KP + kq], b1);
    __syncthreads();
    if (k0 + 32 < K) {  // prefetch next chunk (uniform branch)
      pA0 += 32; pA1 += 32; pB0 += 32; pB1 += 32;
      a0 = *(const float4*)pA0;
      a1 = *(const float4*)pA1;
      b0 = *(const float4*)pB0;
      b1 = *(const float4*)pB1;
    }
    v16h af  = load_a_frag(sA + rowBlk * 16 * KP, KP);
    v16h bf0 = load_b_frag(sB + (colHalf * 32) * KP, KP);
    v16h bf1 = load_b_frag(sB + (colHalf * 32 + 16) * KP, KP);
    acc0 = WMMA_F32_F16(af, bf0, acc0);
    acc1 = WMMA_F32_F16(af, bf1, acc1);
  }
#pragma unroll
  for (int t = 0; t < 2; ++t) {
    const int nG = n0 + colHalf * 32 + t * 16 + nl;
    const float bv = bias ? bias[nG] : 0.f;
    v8f acc = (t == 0) ? acc0 : acc1;
#pragma unroll
    for (int r = 0; r < 8; ++r) {
      const int mG = m0 + rowBlk * 16 + r + 8 * hi;
      C[(long)mG * ldc + nG] = acc[r] + bv;
    }
  }
}

// ---------------------------------------------------------------------------
// Bounds-checked GEMM for small N / K (n1,n2,ei,ej). Per-thread (row,k) fixed
// across iterations -> int offsets, no 64-bit muls in the loop.
// ---------------------------------------------------------------------------
__global__ __launch_bounds__(256) void gemm_small_kernel(
    const float* __restrict__ A, int lda,
    const float* __restrict__ W, int ldw,
    const float* __restrict__ bias,
    float* __restrict__ C, int ldc,
    int M, int N, int K) {
  constexpr int KP = 40;
  __attribute__((aligned(16))) __shared__ _Float16 sA[64 * KP];
  __attribute__((aligned(16))) __shared__ _Float16 sB[64 * KP];
  const int tid = threadIdx.x, lane = tid & 31, w = tid >> 5;
  const int rowBlk = w & 3, colHalf = w >> 2;
  const int m0 = blockIdx.y * 64, n0 = blockIdx.x * 64;
  const int hi = lane >> 4, nl = lane & 15;
  const int kk = tid & 31, rb = tid >> 5;  // idx = tid + t*256 -> r = rb + 8t, k = kk

  int aoff[8]; int woff[8]; bool wok[8];
#pragma unroll
  for (int t = 0; t < 8; ++t) {
    const int r = rb + 8 * t;
    aoff[t] = (m0 + r) * lda + kk;           // gm < M always (M%64==0 in our uses)
    const int gn = n0 + r;
    wok[t] = gn < N;
    woff[t] = (wok[t] ? gn : 0) * ldw + kk;
  }
  v8f acc0 = {}, acc1 = {};
  for (int k0 = 0; k0 < K; k0 += 32) {
    const bool kv = (k0 + kk) < K;
    __syncthreads();
#pragma unroll
    for (int t = 0; t < 8; ++t) {
      const int r = rb + 8 * t;
      sA[r * KP + kk] = (_Float16)(kv ? A[aoff[t] + k0] : 0.f);
      sB[r * KP + kk] = (_Float16)((kv && wok[t]) ? W[woff[t] + k0] : 0.f);
    }
    __syncthreads();
    v16h af  = load_a_frag(sA + rowBlk * 16 * KP, KP);
    v16h bf0 = load_b_frag(sB + (colHalf * 32) * KP, KP);
    v16h bf1 = load_b_frag(sB + (colHalf * 32 + 16) * KP, KP);
    acc0 = WMMA_F32_F16(af, bf0, acc0);
    acc1 = WMMA_F32_F16(af, bf1, acc1);
  }
#pragma unroll
  for (int t = 0; t < 2; ++t) {
    const int nG = n0 + colHalf * 32 + t * 16 + nl;
    if (nG >= N) continue;
    const float bv = bias ? bias[nG] : 0.f;
    v8f acc = (t == 0) ? acc0 : acc1;
#pragma unroll
    for (int r = 0; r < 8; ++r) {
      const int mG = m0 + rowBlk * 16 + r + 8 * hi;
      if (mG < M) C[(long)mG * ldc + nG] = acc[r] + bv;
    }
  }
}

// ---------------------------------------------------------------------------
// gcn[b,i,h,d] = sum_j attn[b,h,i,j] * feature[b,j,h,d]; grid (S/64, B*H).
// A rows contiguous (float4); B staged along contiguous n (float4, coalesced).
// ---------------------------------------------------------------------------
__global__ __launch_bounds__(256) void gemm_attn_feat_kernel(
    const float* __restrict__ attn, const float* __restrict__ feat,
    float* __restrict__ gcnA) {
  constexpr int KP = 40;
  __attribute__((aligned(16))) __shared__ _Float16 sA[64 * KP];
  __attribute__((aligned(16))) __shared__ _Float16 sB[64 * KP];
  const int tid = threadIdx.x, lane = tid & 31, w = tid >> 5;
  const int rowBlk = w & 3, colHalf = w >> 2;
  const int hi = lane >> 4, nl = lane & 15;
  const int z = blockIdx.y, b = z / Hc, h = z % Hc;
  const float* A  = attn + (long)z * Sc * Sc;
  const float* Wb = feat + (long)b * Sc * Dc + h * HDc;  // [k][n] = Wb[k*Dc+n]
  float*       C  = gcnA + (long)b * Sc * Dc + h * HDc;
  const int m0 = blockIdx.x * 64;

  const int rA = tid >> 3, kqA = (tid & 7) * 4;   // A: rows rA, rA+32
  const int kB = tid >> 4, nqB = (tid & 15) * 4;  // B: k = kB, kB+16
  const float* pA0 = A + (long)(m0 + rA) * Sc + kqA;
  const float* pA1 = pA0 + (long)32 * Sc;
  const float* pB0 = Wb + (long)kB * Dc + nqB;
  const float* pB1 = Wb + (long)(kB + 16) * Dc + nqB;

  float4 a0 = *(const float4*)pA0;
  float4 a1 = *(const float4*)pA1;
  float4 b0 = *(const float4*)pB0;
  float4 b1 = *(const float4*)pB1;

  v8f acc0 = {}, acc1 = {};
  for (int k0 = 0; k0 < Sc; k0 += 32) {
    __syncthreads();
    cvt_store4(&sA[rA * KP + kqA], a0);
    cvt_store4(&sA[(rA + 32) * KP + kqA], a1);
#pragma unroll
    for (int c = 0; c < 4; ++c) {
      sB[(nqB + c) * KP + kB]      = (_Float16)((c == 0) ? b0.x : (c == 1) ? b0.y : (c == 2) ? b0.z : b0.w);
      sB[(nqB + c) * KP + kB + 16] = (_Float16)((c == 0) ? b1.x : (c == 1) ? b1.y : (c == 2) ? b1.z : b1.w);
    }
    __syncthreads();
    if (k0 + 32 < Sc) {
      pA0 += 32; pA1 += 32;
      pB0 += (long)32 * Dc; pB1 += (long)32 * Dc;
      a0 = *(const float4*)pA0;
      a1 = *(const float4*)pA1;
      b0 = *(const float4*)pB0;
      b1 = *(const float4*)pB1;
    }
    v16h af  = load_a_frag(sA + rowBlk * 16 * KP, KP);
    v16h bf0 = load_b_frag(sB + (colHalf * 32) * KP, KP);
    v16h bf1 = load_b_frag(sB + (colHalf * 32 + 16) * KP, KP);
    acc0 = WMMA_F32_F16(af, bf0, acc0);
    acc1 = WMMA_F32_F16(af, bf1, acc1);
  }
#pragma unroll
  for (int t = 0; t < 2; ++t) {
    const int nG = colHalf * 32 + t * 16 + nl;
    v8f acc = (t == 0) ? acc0 : acc1;
#pragma unroll
    for (int r = 0; r < 8; ++r) {
      const int mG = m0 + rowBlk * 16 + r + 8 * hi;
      C[(long)mG * Dc + nG] = acc[r];
    }
  }
}

// ---------------------------------------------------------------------------
// Fused edge-MLP + attention logits. 128 threads = 4 waves, 16 rows/wave.
// E=50 staged as 25 aligned float2 pairs + zero pad; mask = bitwise OR of row.
// ---------------------------------------------------------------------------
__global__ __launch_bounds__(128) void edge_mlp_attn_kernel(
    const float* __restrict__ wprob, const float* __restrict__ sloop,
    const float* __restrict__ We1, const float* __restrict__ be1,
    const float* __restrict__ We2, const float* __restrict__ be2,
    const float* __restrict__ srcB, const float* __restrict__ dstB,
    float* __restrict__ attnOut) {
  constexpr int KPa = 72;   // k-stride, K padded to 64
  constexpr int KPh = 136;  // k-stride, K = 128
  __attribute__((aligned(16))) __shared__ _Float16 sW1[128 * KPa];
  __attribute__((aligned(16))) __shared__ _Float16 sW2[16 * KPh];
  __attribute__((aligned(16))) __shared__ _Float16 sA[4][16 * KPa];
  __attribute__((aligned(16))) __shared__ _Float16 sH[4][16 * KPh];
  __shared__ float    sB1[128];
  __shared__ unsigned sMask[4][16];

  const int tid = threadIdx.x, lane = tid & 31, w = tid >> 5;
  const int hi = lane >> 4, nl = lane & 15;

  // stage We1 [128,50] -> sW1[n][k<=64]: 128 rows x 32 pairs
#pragma unroll
  for (int t = 0; t < 32; ++t) {
    const int idx = tid + t * 128;           // 0..4095 pairs
    const int n = idx >> 5, p = idx & 31;
    union { _Float16 h[2]; unsigned u; } pk; pk.u = 0u;
    if (p <= 24) {
      const float2 v = *(const float2*)(We1 + n * Ec + 2 * p);
      pk.h[0] = (_Float16)v.x; pk.h[1] = (_Float16)v.y;
    }
    *(unsigned*)(&sW1[n * KPa + 2 * p]) = pk.u;
  }
  // stage We2 [12,128] -> sW2[n<=16][k=128]: 16 rows x 64 pairs
#pragma unroll
  for (int t = 0; t < 8; ++t) {
    const int idx = tid + t * 128;           // 0..1023 pairs
    const int n = idx >> 6, p = idx & 63;
    union { _Float16 h[2]; unsigned u; } pk; pk.u = 0u;
    if (n < Hc) {
      const float2 v = *(const float2*)(We2 + n * Fc + 2 * p);
      pk.h[0] = (_Float16)v.x; pk.h[1] = (_Float16)v.y;
    }
    *(unsigned*)(&sW2[n * KPh + 2 * p]) = pk.u;
  }
  sB1[tid] = be1[tid];

  const long R = ((long)blockIdx.x * 4 + w) * 16;  // 16 rows share (b,i)
  const int b  = (int)(R / ((long)Sc * Sc));
  const int i  = (int)((R / Sc) % Sc);
  const int j0 = (int)(R % Sc);
  const long rowBase = R * Ec;

  // stage wps rows: lane = pair index (25 valid), t = row; coalesced float2
  {
    const int p = lane;
    const bool pv = (p <= 24);
#pragma unroll
    for (int t = 0; t < 16; ++t) {
      unsigned u = 0u;
      if (pv) {
        const float2 va = *(const float2*)(wprob + rowBase + t * Ec + 2 * p);
        const float2 vb = *(const float2*)(sloop + rowBase + t * Ec + 2 * p);
        union { _Float16 h[2]; unsigned uu; } pk;
        pk.h[0] = (_Float16)(va.x + vb.x);
        pk.h[1] = (_Float16)(va.y + vb.y);
        u = pk.uu;
      }
      *(unsigned*)(&sA[w][t * KPa + 2 * p]) = u;
    }
  }
  __syncthreads();

  // mask: row is masked iff every element is zero -> OR the staged bits
  if (lane < 16) {
    unsigned o = 0u;
#pragma unroll
    for (int p = 0; p < 32; ++p) o |= *(const unsigned*)(&sA[w][lane * KPa + 2 * p]);
    sMask[w][lane] = o;
  }

  // stage 1: hmid = relu(wps @ We1^T + be1) -> sH (16x128 f16)
#pragma unroll
  for (int nt = 0; nt < 8; ++nt) {
    v8f acc = {};
#pragma unroll
    for (int kc = 0; kc < 2; ++kc) {
      v16h a  = load_a_frag(&sA[w][kc * 32], KPa);
      v16h bb = load_b_frag(&sW1[(nt * 16) * KPa + kc * 32], KPa);
      acc = WMMA_F32_F16(a, bb, acc);
    }
    const int n = nt * 16 + nl;
    const float bias = sB1[n];
#pragma unroll
    for (int r = 0; r < 8; ++r) {
      const int m = r + 8 * hi;
      const float v = acc[r] + bias;
      sH[w][m * KPh + n] = (_Float16)((v > 0.f) ? v : 0.f);
    }
  }
  __syncthreads();

  // stage 2: A = hmid @ We2^T -> 16x16 (12 valid heads)
  v8f acc2 = {};
#pragma unroll
  for (int kc = 0; kc < 4; ++kc) {
    v16h a  = load_a_frag(&sH[w][kc * 32], KPh);
    v16h bb = load_b_frag(&sW2[kc * 32], KPh);
    acc2 = WMMA_F32_F16(a, bb, acc2);
  }
  if (nl < Hc) {
    const int h = nl;
    const long bhs = ((long)b * Hc + h) * Sc;
    const float sv = srcB[bhs + i] + be2[h];
    const long outBase = (bhs + i) * Sc;
#pragma unroll
    for (int r = 0; r < 8; ++r) {
      const int m = r + 8 * hi;
      const int j = j0 + m;
      float x = acc2[r] + sv + dstB[bhs + j];
      x = (x > 0.f) ? x : 0.01f * x;            // leaky_relu(0.01)
      if (sMask[w][m] == 0u) x = -INFINITY;
      attnOut[outBase + j] = x;
    }
  }
}

// ---------------------------------------------------------------------------
// Row softmax over j (row length 512), in place. grid = B*H*S, block = 256.
// ---------------------------------------------------------------------------
__global__ __launch_bounds__(256) void softmax_kernel(float* __restrict__ attn) {
  __shared__ float red[256];
  const int tid = threadIdx.x;
  float* p = attn + (long)blockIdx.x * Sc;
  const float v0 = p[tid], v1 = p[tid + 256];
  red[tid] = fmaxf(v0, v1);
  __syncthreads();
  for (int off = 128; off > 0; off >>= 1) {
    if (tid < off) red[tid] = fmaxf(red[tid], red[tid + off]);
    __syncthreads();
  }
  const float mx = red[0];
  __syncthreads();
  const float e0 = __expf(v0 - mx), e1 = __expf(v1 - mx);
  red[tid] = e0 + e1;
  __syncthreads();
  for (int off = 128; off > 0; off >>= 1) {
    if (tid < off) red[tid] += red[tid + off];
    __syncthreads();
  }
  const float inv = 1.f / red[0];
  p[tid] = e0 * inv;
  p[tid + 256] = e1 * inv;
}

// ---------------------------------------------------------------------------
// attn_src / attn_dst: per (b,h,s) 64-length dots with fc_w1 / fc_w2.
// ---------------------------------------------------------------------------
__global__ __launch_bounds__(256) void attn_srcdst_kernel(
    const float* __restrict__ feat, const float* __restrict__ w1,
    const float* __restrict__ w2, float* __restrict__ srcB,
    float* __restrict__ dstB) {
  const int idx = blockIdx.x * 256 + threadIdx.x;
  if (idx >= Bc * Hc * Sc) return;
  const int s = idx % Sc, h = (idx / Sc) % Hc, b = idx / (Hc * Sc);
  const float* f = feat + ((long)(b * Sc + s)) * Dc + h * HDc;
  float a1 = 0.f, a2 = 0.f;
#pragma unroll 8
  for (int d = 0; d < HDc; ++d) {
    const float v = f[d];
    a1 += v * w1[h * HDc + d];
    a2 += v * w2[h * HDc + d];
  }
  srcB[idx] = a1;  // layout (b*H+h)*S + s == idx
  dstB[idx] = a2;
}

// ---------------------------------------------------------------------------
// LayerNorm (unbiased std, /(std+1e-6)) + ReLU -> node. grid = B*S, block 256.
// ---------------------------------------------------------------------------
__global__ __launch_bounds__(256) void layernorm_relu_kernel(
    const float* __restrict__ x, const float* __restrict__ ga,
    const float* __restrict__ be, float* __restrict__ node) {
  __shared__ float red[256];
  const int tid = threadIdx.x;
  const float* xr = x + (long)blockIdx.x * Dc;
  float s = 0.f;
  for (int c = tid; c < Dc; c += 256) s += xr[c];
  red[tid] = s;
  __syncthreads();
  for (int off = 128; off > 0; off >>= 1) {
    if (tid < off) red[tid] += red[tid + off];
    __syncthreads();
  }
  const float mean = red[0] / (float)Dc;
  __syncthreads();
  float s2 = 0.f;
  for (int c = tid; c < Dc; c += 256) { const float d = xr[c] - mean; s2 += d * d; }
  red[tid] = s2;
  __syncthreads();
  for (int off = 128; off > 0; off >>= 1) {
    if (tid < off) red[tid] += red[tid + off];
    __syncthreads();
  }
  const float var = red[0] / (float)(Dc - 1);
  const float inv = 1.f / (sqrtf(fmaxf(var, 0.f)) + 1e-6f);
  for (int c = tid; c < Dc; c += 256) {
    const float v = ga[c] * (xr[c] - mean) * inv + be[c];
    node[(long)blockIdx.x * Dc + c] = (v > 0.f) ? v : 0.f;
  }
}

// ---------------------------------------------------------------------------
// ediag[b,s,e] = weight_adj[b,s,s,e]
// ---------------------------------------------------------------------------
__global__ __launch_bounds__(256) void ediag_kernel(
    const float* __restrict__ wadj, float* __restrict__ ed) {
  const int idx = blockIdx.x * 256 + threadIdx.x;
  if (idx >= Bc * Sc * Ec) return;
  const int e = idx % Ec, s = (idx / Ec) % Sc, b = idx / (Ec * Sc);
  ed[idx] = wadj[(((long)(b * Sc + s)) * Sc + s) * Ec + e];
}

// ---------------------------------------------------------------------------
// P[b,s]=ei+n2 (i-terms), Q[b,s]=ej+n1+bh (j-terms)
// ---------------------------------------------------------------------------
__global__ __launch_bounds__(256) void pq_kernel(
    const float* __restrict__ ei, const float* __restrict__ ej,
    const float* __restrict__ n1, const float* __restrict__ n2,
    const float* __restrict__ bh, float* __restrict__ P,
    float* __restrict__ Q) {
  const int idx = blockIdx.x * 256 + threadIdx.x;
  if (idx >= Bc * Sc * DEc) return;
  const int e = idx % DEc;
  P[idx] = ei[idx] + n2[idx];
  Q[idx] = ej[idx] + n1[idx] + bh[e];
}

// ---------------------------------------------------------------------------
// edge_out = wadj @ Wh_e^T + P[b,i] + Q[b,j]. 256 threads = 8 waves x 16 rows.
// ---------------------------------------------------------------------------
__global__ __launch_bounds__(256) void edge_out_kernel(
    const float* __restrict__ wadj, const float* __restrict__ Whe,
    const float* __restrict__ P, const float* __restrict__ Q,
    float* __restrict__ out) {
  constexpr int KPa = 72;
  __attribute__((aligned(16))) __shared__ _Float16 sW[64 * KPa];
  __attribute__((aligned(16))) __shared__ _Float16 sA[8][16 * KPa];
  const int tid = threadIdx.x, lane = tid & 31, w = tid >> 5;
  const int hi = lane >> 4, nl = lane & 15;

  // stage Wh_e [50,50] -> sW[n<=64][k<=64]: 64 rows x 32 pairs
#pragma unroll
  for (int t = 0; t < 8; ++t) {
    const int idx = tid + t * 256;           // 0..2047 pairs
    const int n = idx >> 5, p = idx & 31;
    union { _Float16 h[2]; unsigned u; } pk; pk.u = 0u;
    if (n < DEc && p <= 24) {
      const float2 v = *(const float2*)(Whe + n * Ec + 2 * p);
      pk.h[0] = (_Float16)v.x; pk.h[1] = (_Float16)v.y;
    }
    *(unsigned*)(&sW[n * KPa + 2 * p]) = pk.u;
  }

  const long R = ((long)blockIdx.x * 8 + w) * 16;
  const int b  = (int)(R / ((long)Sc * Sc));
  const int i  = (int)((R / Sc) % Sc);
  const int j0 = (int)(R % Sc);
  const long rowBase = R * Ec;

  {
    const int p = lane;
    const bool pv = (p <= 24);
#pragma unroll
    for (int t = 0; t < 16; ++t) {
      unsigned u = 0u;
      if (pv) {
        const float2 v = *(const float2*)(wadj + rowBase + t * Ec + 2 * p);
        union { _Float16 h[2]; unsigned uu; } pk;
        pk.h[0] = (_Float16)v.x; pk.h[1] = (_Float16)v.y;
        u = pk.uu;
      }
      *(unsigned*)(&sA[w][t * KPa + 2 * p]) = u;
    }
  }
  __syncthreads();

  v8f acc[4];
  v8f zero = {};
#pragma unroll
  for (int nt = 0; nt < 4; ++nt) acc[nt] = zero;
#pragma unroll
  for (int kc = 0; kc < 2; ++kc) {
    v16h a = load_a_frag(&sA[w][kc * 32], KPa);
#pragma unroll
    for (int nt = 0; nt < 4; ++nt) {
      v16h bb = load_b_frag(&sW[(nt * 16) * KPa + kc * 32], KPa);
      acc[nt] = WMMA_F32_F16(a, bb, acc[nt]);
    }
  }
#pragma unroll
  for (int nt = 0; nt < 4; ++nt) {
    const int n = nt * 16 + nl;
    if (n >= DEc) continue;
    const float pv = P[((long)b * Sc + i) * DEc + n];
#pragma unroll
    for (int r = 0; r < 8; ++r) {
      const int m = r + 8 * hi;
      const int j = j0 + m;
      out[(R + m) * (long)DEc + n] = acc[nt][r] + pv + Q[((long)b * Sc + j) * DEc + n];
    }
  }
}

// ---------------------------------------------------------------------------
extern "C" void kernel_launch(void* const* d_in, const int* in_sizes, int n_in,
                              void* d_out, int out_size, void* d_ws, size_t ws_size,
                              hipStream_t stream) {
  const float* wprob  = (const float*)d_in[0];
  const float* wadj   = (const float*)d_in[1];
  const float* gcn_in = (const float*)d_in[2];
  const float* sloop  = (const float*)d_in[3];
  const float* W_lin  = (const float*)d_in[4];
  const float* b_lin  = (const float*)d_in[5];
  const float* fc_w1  = (const float*)d_in[6];
  const float* fc_w2  = (const float*)d_in[7];
  const float* We1    = (const float*)d_in[8];
  const float* be1    = (const float*)d_in[9];
  const float* We2    = (const float*)d_in[10];
  const float* be2    = (const float*)d_in[11];
  const float* W_out  = (const float*)d_in[12];
  const float* b_out  = (const float*)d_in[13];
  const float* ln_a   = (const float*)d_in[14];
  const float* ln_b   = (const float*)d_in[15];
  const float* Wh_e   = (const float*)d_in[16];
  const float* Wh_ei  = (const float*)d_in[17];
  const float* Wh_ej  = (const float*)d_in[18];
  const float* Wh_n1  = (const float*)d_in[19];
  const float* Wh_n2  = (const float*)d_in[20];
  const float* bh     = (const float*)d_in[21];

  float* ws     = (float*)d_ws;
  float* feat   = ws + FEAT_OFF;
  float* attnB  = ws + ATTN_OFF;
  float* gcnA   = ws + GCNA_OFF;
  float* gcnP   = ws + GCNP_OFF;
  float* srcB   = ws + SRC_OFF;
  float* dstB   = ws + DST_OFF;
  float* ediagB = ws + EDIAG_OFF;
  float* n1B    = ws + N1_OFF;
  float* n2B    = ws + N2_OFF;
  float* eiB    = ws + EI_OFF;
  float* ejB    = ws + EJ_OFF;
  float* Pb     = ws + P_OFF;
  float* Qb     = ws + Q_OFF;

  float* nodeOut = (float*)d_out;                       // [B,S,D]
  float* edgeOut = (float*)d_out + (long)Bc * Sc * Dc;  // [B,S,S,DE]

  // 1) feature = gcn_inputs @ W_lin^T + b_lin  (1024x768x768, fully aligned)
  gemm_full_kernel<<<dim3(Dc / 64, (Bc * Sc) / 64), 256, 0, stream>>>(
      gcn_in, Dc, W_lin, Dc, b_lin, feat, Dc, Dc);

  // 2) attn_src / attn_dst
  attn_srcdst_kernel<<<dim3((Bc * Hc * Sc + 255) / 256), 256, 0, stream>>>(
      feat, fc_w1, fc_w2, srcB, dstB);

  // 3) fused edge MLP + logits (hmid never hits HBM)
  edge_mlp_attn_kernel<<<dim3((Bc * Sc * Sc) / 64), 128, 0, stream>>>(
      wprob, sloop, We1, be1, We2, be2, srcB, dstB, attnB);

  // 4) softmax over j, in place
  softmax_kernel<<<dim3(Bc * Hc * Sc), 256, 0, stream>>>(attnB);

  // 5) gcn = attn @ feature  (per b,h: 512x512x64)
  gemm_attn_feat_kernel<<<dim3(Sc / 64, Bc * Hc), 256, 0, stream>>>(attnB, feat, gcnA);

  // 6) gcn @ W_out^T + b_out  (fully aligned)
  gemm_full_kernel<<<dim3(Dc / 64, (Bc * Sc) / 64), 256, 0, stream>>>(
      gcnA, Dc, W_out, Dc, b_out, gcnP, Dc, Dc);

  // 7) LayerNorm (ddof=1) + ReLU -> node (first output)
  layernorm_relu_kernel<<<dim3(Bc * Sc), 256, 0, stream>>>(gcnP, ln_a, ln_b, nodeOut);

  // 8) diagonal of weight_adj
  ediag_kernel<<<dim3((Bc * Sc * Ec + 255) / 256), 256, 0, stream>>>(wadj, ediagB);

  // 9) small projections (WMMA): n1,n2 (K=768), ei,ej (K=50)
  gemm_small_kernel<<<dim3(1, (Bc * Sc) / 64), 256, 0, stream>>>(
      nodeOut, Dc, Wh_n1, Dc, nullptr, n1B, DEc, Bc * Sc, DEc, Dc);
  gemm_small_kernel<<<dim3(1, (Bc * Sc) / 64), 256, 0, stream>>>(
      nodeOut, Dc, Wh_n2, Dc, nullptr, n2B, DEc, Bc * Sc, DEc, Dc);
  gemm_small_kernel<<<dim3(1, (Bc * Sc) / 64), 256, 0, stream>>>(
      ediagB, Ec, Wh_ei, Ec, nullptr, eiB, DEc, Bc * Sc, DEc, Ec);
  gemm_small_kernel<<<dim3(1, (Bc * Sc) / 64), 256, 0, stream>>>(
      ediagB, Ec, Wh_ej, Ec, nullptr, ejB, DEc, Bc * Sc, DEc, Ec);

  // 10) combine broadcast terms
  pq_kernel<<<dim3((Bc * Sc * DEc + 255) / 256), 256, 0, stream>>>(
      eiB, ejB, n1B, n2B, bh, Pb, Qb);

  // 11) edge_out = wadj @ Wh_e^T + P[b,i] + Q[b,j] (second output)
  edge_out_kernel<<<dim3((Bc * Sc * Sc) / 128), 256, 0, stream>>>(
      wadj, Wh_e, Pb, Qb, edgeOut);
}
